// GS4_fuse_67319317397628
// MI455X (gfx1250) — matrix operand order repaired and checked by reading, not compiled, not run on hardware
//
#include <hip/hip_runtime.h>
#include <math.h>

// ---------------- CDNA5 WMMA types ----------------
typedef __attribute__((ext_vector_type(16))) _Float16 v16h;
typedef __attribute__((ext_vector_type(8)))  float    v8f;

#define S_VIEWS 4
#define CH      23          // 19 feat + 4 dir
#define FC      19
#define NPTS    320000
#define HWPIX   40000
#define DDEPTH  8
#define HH      200
#define WW      200

#define PTS_PER_BLK 32
#define ROWS        128     // PTS_PER_BLK * S_VIEWS
#define K64         64

// CDNA5 async global->LDS path (guarded so the file compiles on either toolchain)
#if defined(__has_builtin)
#if __has_builtin(__builtin_amdgcn_global_load_async_to_lds_b32)
#define USE_ASYNC_LDS 1
#endif
#endif

// ---- fragment loaders per CDNA5 ISA §7.12.2 (wave32) ----
// A 16x32 f16: lane<16 row=lane, halfs = K{kb..kb+7, kb+16..kb+23}, kb=0; lane>=16 kb=8
__device__ __forceinline__ v16h load_fragA(const _Float16* base, int mtile, int ktile, int lane) {
  const _Float16* p = base + (size_t)(mtile * 16 + (lane & 15)) * K64 + ktile * 32 + ((lane & 16) ? 8 : 0);
  v16h f;
#pragma unroll
  for (int i = 0; i < 8; ++i) { f[i] = p[i]; f[i + 8] = p[i + 16]; }
  return f;
}
// B 32x16 f16 (stored in LDS as [col][K] row-major): lane<16 col=lane, K{kb..kb+7, kb+16..kb+23}, kb=0; lane>=16 kb=8
__device__ __forceinline__ v16h load_fragB(const _Float16* base, int ntile, int ktile, int lane) {
  const _Float16* p = base + (size_t)(ntile * 16 + (lane & 15)) * K64 + ktile * 32 + ((lane & 16) ? 8 : 0);
  v16h f;
#pragma unroll
  for (int i = 0; i < 8; ++i) { f[i] = p[i]; f[i + 8] = p[i + 16]; }
  return f;
}

// one 16x16 output tile, K=64 (2 WMMAs), bias+relu on writeback into sD[ROWS][32]
__device__ __forceinline__ void wmma_tile(const _Float16* sA, const _Float16* sB, float* sD,
                                          int mtile, int ntile, const float* __restrict__ bias,
                                          int nvalid, int lane) {
  int col  = lane & 15;
  int ocol = ntile * 16 + col;
  float bv = (ocol < nvalid) ? bias[ocol] : 0.f;
  v8f c;
#pragma unroll
  for (int i = 0; i < 8; ++i) c[i] = bv;
  v16h a0 = load_fragA(sA, mtile, 0, lane);
  v16h b0 = load_fragB(sB, ntile, 0, lane);
  c = __builtin_amdgcn_wmma_f32_16x16x32_f16(false, a0, false, b0, (short)0, c, false, false);
  v16h a1 = load_fragA(sA, mtile, 1, lane);
  v16h b1 = load_fragB(sB, ntile, 1, lane);
  c = __builtin_amdgcn_wmma_f32_16x16x32_f16(false, a1, false, b1, (short)0, c, false, false);
  int rbase = mtile * 16 + ((lane & 16) ? 8 : 0);
#pragma unroll
  for (int i = 0; i < 8; ++i) {
    float v = c[i]; v = v > 0.f ? v : 0.f;
    sD[(size_t)(rbase + i) * 32 + ocol] = v;
  }
}

__device__ __forceinline__ float dotN(const float* __restrict__ w, const float* v, int n) {
  float a = 0.f;
  for (int i = 0; i < n; ++i) a += w[i] * v[i];
  return a;
}
__device__ __forceinline__ float softplusf(float z) { return (z > 20.f) ? z : log1pf(expf(z)); }
__device__ __forceinline__ float sigmoidf_(float z) { return 1.f / (1.f + expf(-z)); }

// =================== Kernel 1: per-point fused aggregation + NeRF heads ===================
__global__ __launch_bounds__(256)
void k_point(const float* __restrict__ vox, const float* __restrict__ img,
             const float* __restrict__ w_view, const float* __restrict__ b_view,
             const float* __restrict__ w_glob, const float* __restrict__ b_glob,
             const float* __restrict__ w_aggw, const float* __restrict__ b_aggw,
             const float* __restrict__ w_fc,   const float* __restrict__ b_fc,
             const float* __restrict__ w_lr0,  const float* __restrict__ b_lr0,
             const float* __restrict__ w_sig,  const float* __restrict__ b_sig,
             const float* __restrict__ w_c1,   const float* __restrict__ b_c1,
             const float* __restrict__ w_c2,   const float* __restrict__ b_c2,
             float* __restrict__ out_rad, float* __restrict__ out_sig, float* __restrict__ out_fnerf,
             float* __restrict__ ws_x, float* __restrict__ ws_sig, float* __restrict__ ws_rad) {
  __shared__ float s_raw[ROWS][24];                       // original 23 channels
  __shared__ union { float ifr[ROWS][20]; float D[ROWS][32]; } u;  // phase-overlapped
  __shared__ _Float16 s_A[ROWS][K64];
  __shared__ _Float16 s_B[32][K64];
  __shared__ float s_logit[ROWS];
  __shared__ float s_imf[PTS_PER_BLK][32];
  __shared__ float s_if16[PTS_PER_BLK][16];
  __shared__ float s_x[PTS_PER_BLK][24];

  const int tid  = threadIdx.x;
  const int lane = tid & 31;
  const int wave = tid >> 5;
  const int p0   = blockIdx.x * PTS_PER_BLK;
  const size_t gbase = (size_t)p0 * S_VIEWS;              // global row base

  // ---- load raw img_feat_rgb_dir into LDS ----
#ifdef USE_ASYNC_LDS
  {
    typedef __attribute__((address_space(1))) int gint_t;  // global int* per builtin signature
    typedef __attribute__((address_space(3))) int lint_t;  // LDS int*
    for (int i = tid; i < ROWS * CH; i += 256) {
      __builtin_amdgcn_global_load_async_to_lds_b32(
          (gint_t*)(void*)(img + gbase * CH + i),
          (lint_t*)(void*)&s_raw[i / CH][i % CH], 0, 0);
    }
#if __has_builtin(__builtin_amdgcn_s_wait_asynccnt)
    __builtin_amdgcn_s_wait_asynccnt(0);
#else
    asm volatile("s_wait_asynccnt 0" ::: "memory");
#endif
  }
#else
  for (int i = tid; i < ROWS * CH; i += 256) {
    s_raw[i / CH][i % CH] = img[gbase * CH + i];
  }
#endif
  __syncthreads();

  // ---- view_feat = relu(dir @ Wv^T + bv); ifr = feat + view_feat ----
  for (int i = tid; i < ROWS * FC; i += 256) {
    int r = i / FC, c = i % FC;
    float vf = b_view[c];
#pragma unroll
    for (int j = 0; j < 4; ++j) vf += w_view[c * 4 + j] * s_raw[r][FC + j];
    vf = vf > 0.f ? vf : 0.f;
    u.ifr[r][c] = s_raw[r][c] + vf;
  }
  __syncthreads();

  // ---- var (ddof=1) / mean over views; stage A-tile f16 [ifr | var | mean | 0-pad] ----
  for (int i = tid; i < PTS_PER_BLK * FC; i += 256) {
    int p = i / FC, c = i % FC;
    float x0 = u.ifr[4 * p + 0][c], x1 = u.ifr[4 * p + 1][c];
    float x2 = u.ifr[4 * p + 2][c], x3 = u.ifr[4 * p + 3][c];
    float m  = 0.25f * (x0 + x1 + x2 + x3);
    float ss = x0 * x0 + x1 * x1 + x2 * x2 + x3 * x3;
    float var = (ss - 4.f * m * m) * (1.f / 3.f);
#pragma unroll
    for (int s = 0; s < 4; ++s) {
      int r = 4 * p + s;
      s_A[r][c]           = (_Float16)u.ifr[r][c];
      s_A[r][FC + c]      = (_Float16)var;
      s_A[r][2 * FC + c]  = (_Float16)m;
    }
  }
  for (int i = tid; i < ROWS * (K64 - 3 * FC); i += 256)      // zero K=57..63
    s_A[i / 7][57 + (i % 7)] = (_Float16)0.f;
  for (int i = tid; i < 32 * K64; i += 256) {                  // B tile: agg_global_w [32][57]
    int o = i / K64, k = i % K64;
    s_B[o][k] = (_Float16)((k < 57) ? w_glob[o * 57 + k] : 0.f);
  }
  __syncthreads();

  // ---- GEMM1: global_feat = relu(A[128x57] x W^T[57x32] + b) via WMMA ----
#pragma unroll
  for (int nt = 0; nt < 2; ++nt)
    wmma_tile(&s_A[0][0], &s_B[0][0], &u.D[0][0], wave, nt, b_glob, 32, lane);
  __syncthreads();

  // ---- agg_w logits ----
  if (tid < ROWS) {
    float lg = b_aggw[0] + dotN(w_aggw, &u.D[tid][0], 32);
    s_logit[tid] = lg > 0.f ? lg : 0.f;
  }
  __syncthreads();

  // ---- softmax over views + im_feat ----
  for (int i = tid; i < PTS_PER_BLK * 32; i += 256) {
    int p = i >> 5, c = i & 31;
    float l0 = s_logit[4 * p], l1 = s_logit[4 * p + 1], l2 = s_logit[4 * p + 2], l3 = s_logit[4 * p + 3];
    float m = fmaxf(fmaxf(l0, l1), fmaxf(l2, l3));
    float e0 = expf(l0 - m), e1 = expf(l1 - m), e2 = expf(l2 - m), e3 = expf(l3 - m);
    float inv = 1.f / (e0 + e1 + e2 + e3);
    s_imf[p][c] = (u.D[4 * p][c] * e0 + u.D[4 * p + 1][c] * e1 +
                   u.D[4 * p + 2][c] * e2 + u.D[4 * p + 3][c] * e3) * inv;
  }
  __syncthreads();

  // ---- agg_fc: 32 -> 16, relu ----
  for (int i = tid; i < PTS_PER_BLK * 16; i += 256) {
    int p = i >> 4, o = i & 15;
    float v = b_fc[o] + dotN(&w_fc[o * 32], &s_imf[p][0], 32);
    s_if16[p][o] = v > 0.f ? v : 0.f;
  }
  __syncthreads();

  // ---- feature_nerf + lr0 (24 -> 24, relu) ----
  for (int i = tid; i < PTS_PER_BLK * 24; i += 256) {
    int p = i / 24, o = i % 24, gp = p0 + p;
    float fn = (o < 8) ? vox[(size_t)gp * 8 + o] : s_if16[p][o - 8];
    __builtin_nontemporal_store(fn, &out_fnerf[(size_t)gp * 24 + o]);  // stream-only output
    float v = b_lr0[o];
#pragma unroll
    for (int j = 0; j < 8; ++j)  v += w_lr0[o * 24 + j] * vox[(size_t)gp * 8 + j];
#pragma unroll
    for (int j = 0; j < 16; ++j) v += w_lr0[o * 24 + 8 + j] * s_if16[p][j];
    v = v > 0.f ? v : 0.f;
    s_x[p][o] = v;
    ws_x[(size_t)gp * 24 + o] = v;   // re-read by k_composite: keep cacheable (192MB L2)
  }
  __syncthreads();

  // ---- sigma = softplus(x @ w + b) ----
  if (tid < PTS_PER_BLK) {
    int gp = p0 + tid;
    float z = b_sig[0] + dotN(w_sig, &s_x[tid][0], 24);
    float sp = softplusf(z);
    out_sig[gp] = sp; ws_sig[gp] = sp;
  }

  // ---- stage A/B for color1: rows=[x(24)|raw(23)|0], W[24][47] padded ----
  for (int i = tid; i < ROWS * K64; i += 256) {
    int r = i / K64, k = i % K64, p = r >> 2;
    float v = 0.f;
    if (k < 24) v = s_x[p][k];
    else if (k < 47) v = s_raw[r][k - 24];
    s_A[r][k] = (_Float16)v;
  }
  for (int i = tid; i < 32 * K64; i += 256) {
    int o = i / K64, k = i % K64;
    s_B[o][k] = (_Float16)((o < 24 && k < 47) ? w_c1[o * 47 + k] : 0.f);
  }
  __syncthreads();

  // ---- GEMM2: color1 via WMMA ----
#pragma unroll
  for (int nt = 0; nt < 2; ++nt)
    wmma_tile(&s_A[0][0], &s_B[0][0], &u.D[0][0], wave, nt, b_c1, 24, lane);
  __syncthreads();

  // ---- color2 logits ----
  if (tid < ROWS) {
    float lg = b_c2[0] + dotN(w_c2, &u.D[tid][0], 24);
    s_logit[tid] = lg > 0.f ? lg : 0.f;
  }
  __syncthreads();

  // ---- radiance = sum_views softmax(logit) * rgb ----
  if (tid < PTS_PER_BLK * 3) {
    int p = tid / 3, c = tid % 3, gp = p0 + p;
    float l0 = s_logit[4 * p], l1 = s_logit[4 * p + 1], l2 = s_logit[4 * p + 2], l3 = s_logit[4 * p + 3];
    float m = fmaxf(fmaxf(l0, l1), fmaxf(l2, l3));
    float e0 = expf(l0 - m), e1 = expf(l1 - m), e2 = expf(l2 - m), e3 = expf(l3 - m);
    float inv = 1.f / (e0 + e1 + e2 + e3);
    float rad = (s_raw[4 * p][16 + c] * e0 + s_raw[4 * p + 1][16 + c] * e1 +
                 s_raw[4 * p + 2][16 + c] * e2 + s_raw[4 * p + 3][16 + c] * e3) * inv;
    out_rad[(size_t)gp * 3 + c] = rad;
    ws_rad[(size_t)gp * 3 + c] = rad;
  }
}

// =================== Kernel 2: alpha compositing + rot fill ===================
__global__ __launch_bounds__(256)
void k_composite(const float* __restrict__ ws_sig, const float* __restrict__ ws_rad,
                 const float* __restrict__ ws_x, float* __restrict__ ws_xw,
                 float* __restrict__ ws_rgbvr, float* __restrict__ out_rot) {
  int pix = blockIdx.x * blockDim.x + threadIdx.x;
  if (pix >= HWPIX) return;
  float T = 1.f, ar = 0.f, ag = 0.f, ab = 0.f;
  float ax[24];
#pragma unroll
  for (int c = 0; c < 24; ++c) ax[c] = 0.f;
  for (int dd = 0; dd < DDEPTH; ++dd) {
    int pt = pix * DDEPTH + dd;
    float a = 1.f - expf(-ws_sig[pt]);
    float w = a * T;
    T *= (1.f - a + 1e-10f);
    ar += w * ws_rad[(size_t)pt * 3 + 0];
    ag += w * ws_rad[(size_t)pt * 3 + 1];
    ab += w * ws_rad[(size_t)pt * 3 + 2];
#pragma unroll
    for (int c = 0; c < 24; ++c) ax[c] += w * ws_x[(size_t)pt * 24 + c];
  }
  ws_rgbvr[(size_t)pix * 3 + 0] = ar;
  ws_rgbvr[(size_t)pix * 3 + 1] = ag;
  ws_rgbvr[(size_t)pix * 3 + 2] = ab;
#pragma unroll
  for (int c = 0; c < 24; ++c) ws_xw[(size_t)pix * 24 + c] = ax[c];
#pragma unroll
  for (int c = 0; c < 4; ++c)
    __builtin_nontemporal_store(0.5f, &out_rot[(size_t)pix * 4 + c]);
}

// =================== Kernels 3/4: 3x3 SAME conv + relu (HWIO weights) ===================
template <int CIN, int COUT>
__global__ __launch_bounds__(256)
void k_conv3x3_relu(const float* __restrict__ in, const float* __restrict__ w,
                    const float* __restrict__ b, float* __restrict__ out) {
  int pix = blockIdx.x * blockDim.x + threadIdx.x;
  if (pix >= HWPIX) return;
  int y = pix / WW, x = pix % WW;
  float acc[COUT];
#pragma unroll
  for (int o = 0; o < COUT; ++o) acc[o] = b[o];
  for (int ky = 0; ky < 3; ++ky) {
    int yy = y + ky - 1; if (yy < 0 || yy >= HH) continue;
    for (int kx = 0; kx < 3; ++kx) {
      int xx = x + kx - 1; if (xx < 0 || xx >= WW) continue;
      const float* ip = in + (size_t)(yy * WW + xx) * CIN;
      const float* wp = w + (size_t)((ky * 3 + kx) * CIN) * COUT;  // uniform -> scalar loads
      for (int ic = 0; ic < CIN; ++ic) {
        float iv = ip[ic];
#pragma unroll
        for (int o = 0; o < COUT; ++o) acc[o] += iv * wp[ic * COUT + o];
      }
    }
  }
#pragma unroll
  for (int o = 0; o < COUT; ++o) {
    float v = acc[o];
    out[(size_t)pix * COUT + o] = v > 0.f ? v : 0.f;
  }
}

// =================== Kernel 5: GS heads ===================
__global__ __launch_bounds__(256)
void k_gs_heads(const float* __restrict__ xg, const float* __restrict__ rgbvr,
                const float* __restrict__ w_sc1, const float* __restrict__ b_sc1,
                const float* __restrict__ w_sc2, const float* __restrict__ b_sc2,
                const float* __restrict__ w_op1, const float* __restrict__ b_op1,
                const float* __restrict__ w_op2, const float* __restrict__ b_op2,
                const float* __restrict__ w_cg1, const float* __restrict__ b_cg1,
                const float* __restrict__ w_cg2, const float* __restrict__ b_cg2,
                float* __restrict__ out_scale, float* __restrict__ out_opa,
                float* __restrict__ out_color) {
  int pix = blockIdx.x * blockDim.x + threadIdx.x;
  if (pix >= HWPIX) return;
  float xv[27];
#pragma unroll
  for (int j = 0; j < 24; ++j) xv[j] = xg[(size_t)pix * 24 + j];
#pragma unroll
  for (int c = 0; c < 3; ++c) xv[24 + c] = rgbvr[(size_t)pix * 3 + c];
  float h[24];
  // scale
#pragma unroll
  for (int o = 0; o < 24; ++o) {
    float v = b_sc1[o] + dotN(&w_sc1[o * 24], xv, 24);
    h[o] = v > 0.f ? v : 0.f;
  }
#pragma unroll
  for (int c = 0; c < 3; ++c)
    __builtin_nontemporal_store(softplusf(b_sc2[c] + dotN(&w_sc2[c * 24], h, 24)),
                                &out_scale[(size_t)pix * 3 + c]);
  // opacity
#pragma unroll
  for (int o = 0; o < 24; ++o) {
    float v = b_op1[o] + dotN(&w_op1[o * 24], xv, 24);
    h[o] = v > 0.f ? v : 0.f;
  }
  __builtin_nontemporal_store(sigmoidf_(b_op2[0] + dotN(w_op2, h, 24)), &out_opa[pix]);
  // color (input = [xg, rgb_vr], 27)
#pragma unroll
  for (int o = 0; o < 24; ++o) {
    float v = b_cg1[o] + dotN(&w_cg1[o * 27], xv, 27);
    h[o] = v > 0.f ? v : 0.f;
  }
#pragma unroll
  for (int c = 0; c < 3; ++c)
    __builtin_nontemporal_store(sigmoidf_(b_cg2[c] + dotN(&w_cg2[c * 24], h, 24)),
                                &out_color[(size_t)pix * 3 + c]);
}

// =================== launch ===================
extern "C" void kernel_launch(void* const* d_in, const int* in_sizes, int n_in,
                              void* d_out, int out_size, void* d_ws, size_t ws_size,
                              hipStream_t stream) {
  const float* vox = (const float*)d_in[0];
  const float* img = (const float*)d_in[1];
  // d_in[2] = z_vals (unused by the reference forward)
  const float* w_view = (const float*)d_in[3];  const float* b_view = (const float*)d_in[4];
  const float* w_glob = (const float*)d_in[5];  const float* b_glob = (const float*)d_in[6];
  const float* w_aggw = (const float*)d_in[7];  const float* b_aggw = (const float*)d_in[8];
  const float* w_fc   = (const float*)d_in[9];  const float* b_fc   = (const float*)d_in[10];
  const float* w_lr0  = (const float*)d_in[11]; const float* b_lr0  = (const float*)d_in[12];
  const float* w_sig  = (const float*)d_in[13]; const float* b_sig  = (const float*)d_in[14];
  const float* w_c1   = (const float*)d_in[15]; const float* b_c1   = (const float*)d_in[16];
  const float* w_c2   = (const float*)d_in[17]; const float* b_c2   = (const float*)d_in[18];
  const float* w_op1  = (const float*)d_in[19]; const float* b_op1  = (const float*)d_in[20];
  const float* w_op2  = (const float*)d_in[21]; const float* b_op2  = (const float*)d_in[22];
  const float* w_sc1  = (const float*)d_in[23]; const float* b_sc1  = (const float*)d_in[24];
  const float* w_sc2  = (const float*)d_in[25]; const float* b_sc2  = (const float*)d_in[26];
  const float* w_cg1  = (const float*)d_in[27]; const float* b_cg1  = (const float*)d_in[28];
  const float* w_cg2  = (const float*)d_in[29]; const float* b_cg2  = (const float*)d_in[30];
  const float* w_u1   = (const float*)d_in[31]; const float* b_u1   = (const float*)d_in[32];
  const float* w_u2   = (const float*)d_in[33]; const float* b_u2   = (const float*)d_in[34];

  float* out = (float*)d_out;
  float* out_rad   = out;                                   // [N,3]
  float* out_sig   = out + (size_t)NPTS * 3;                // [N,1]
  float* out_rot   = out_sig + NPTS;                        // [HW,4]
  float* out_scale = out_rot + (size_t)HWPIX * 4;           // [HW,3]
  float* out_opa   = out_scale + (size_t)HWPIX * 3;         // [HW,1]
  float* out_color = out_opa + HWPIX;                       // [HW,3]
  float* out_fnerf = out_color + (size_t)HWPIX * 3;         // [N,24]
  float* out_fgs   = out_fnerf + (size_t)NPTS * 24;         // [HW,24]

  float* ws = (float*)d_ws;
  float* ws_x     = ws;                                     // [N,24]
  float* ws_sig   = ws_x + (size_t)NPTS * 24;               // [N]
  float* ws_rad   = ws_sig + NPTS;                          // [N,3]
  float* ws_xw    = ws_rad + (size_t)NPTS * 3;              // [HW,24]
  float* ws_rgbvr = ws_xw + (size_t)HWPIX * 24;             // [HW,3]
  float* ws_h     = ws_rgbvr + (size_t)HWPIX * 3;           // [HW,16]

  k_point<<<NPTS / PTS_PER_BLK, 256, 0, stream>>>(
      vox, img, w_view, b_view, w_glob, b_glob, w_aggw, b_aggw, w_fc, b_fc,
      w_lr0, b_lr0, w_sig, b_sig, w_c1, b_c1, w_c2, b_c2,
      out_rad, out_sig, out_fnerf, ws_x, ws_sig, ws_rad);

  int pgrid = (HWPIX + 255) / 256;
  k_composite<<<pgrid, 256, 0, stream>>>(ws_sig, ws_rad, ws_x, ws_xw, ws_rgbvr, out_rot);
  k_conv3x3_relu<24, 16><<<pgrid, 256, 0, stream>>>(ws_xw, w_u1, b_u1, ws_h);
  k_conv3x3_relu<16, 24><<<pgrid, 256, 0, stream>>>(ws_h, w_u2, b_u2, out_fgs);
  k_gs_heads<<<pgrid, 256, 0, stream>>>(out_fgs, ws_rgbvr,
      w_sc1, b_sc1, w_sc2, b_sc2, w_op1, b_op1, w_op2, b_op2,
      w_cg1, b_cg1, w_cg2, b_cg2, out_scale, out_opa, out_color);
}
